// ConcatProductAttention_86174223827277
// MI455X (gfx1250) — matrix operand order repaired and checked by reading, not compiled, not run on hardware
//
#include <hip/hip_runtime.h>
#include <hip/hip_bf16.h>
#include <math.h>

#define B_ 8
#define L_ 2048
#define D_ 1024

typedef __attribute__((ext_vector_type(16))) __bf16 v16bf;
typedef __attribute__((ext_vector_type(8)))  __bf16 v8bf;
typedef __attribute__((ext_vector_type(8)))  float  v8f;

// ---------------------------------------------------------------------------
// CDNA5 async global->LDS copy (16B per lane), tracked by ASYNCcnt.
// ---------------------------------------------------------------------------
__device__ __forceinline__ void async_copy_b128(unsigned int lds_addr, const void* gaddr) {
  asm volatile("global_load_async_to_lds_b128 %0, %1, off"
               :: "v"(lds_addr), "v"(gaddr) : "memory");
}

// ---------------------------------------------------------------------------
// Kernel 1: e[b,j] = exp( dot(k[b,j,:], W2) + b2 )
// One wave32 per row; 8 rows per 256-thread block.
// (q_s is constant along the softmax axis and cancels: q/W1/b1 never needed.)
// ---------------------------------------------------------------------------
__global__ __launch_bounds__(256)
void proj_exp_kernel(const float* __restrict__ k,
                     const float* __restrict__ W2,
                     const float* __restrict__ b2,
                     float* __restrict__ e) {
  const int lane = threadIdx.x & 31;
  const int wid  = threadIdx.x >> 5;
  const int row  = blockIdx.x * 8 + wid;          // [0, B*L)
  const float4* kr = (const float4*)(k + (size_t)row * D_);
  const float4* w4 = (const float4*)W2;
  float acc = 0.f;
#pragma unroll
  for (int t = 0; t < D_ / 4 / 32; ++t) {         // 8 float4 per lane
    float4 a = kr[lane + t * 32];
    float4 w = w4[lane + t * 32];
    acc += a.x * w.x + a.y * w.y + a.z * w.z + a.w * w.w;
  }
#pragma unroll
  for (int off = 16; off > 0; off >>= 1)
    acc += __shfl_xor(acc, off, 32);
  if (lane == 0) e[row] = expf(acc + b2[0]);
}

// ---------------------------------------------------------------------------
// Kernel 2: per row (b,i):  Z = sum_j unmask * e[b,j]
//           attn[b,i,j] = unmask ? e[b,j]/Z : 0   (f32 output)
//           optionally also a bf16 copy for the WMMA GEMM (abf != nullptr)
// ---------------------------------------------------------------------------
__global__ __launch_bounds__(256)
void softmax_rows_kernel(const unsigned char* __restrict__ mask,
                         const float* __restrict__ e,
                         float* __restrict__ attn,
                         __bf16* __restrict__ abf) {
  const int row = blockIdx.x;                     // b*L + i
  const int b   = row >> 11;                      // L = 2048
  const unsigned char* mrow = mask + (size_t)row * L_;
  const float* eb = e + ((size_t)b << 11);
  float* arow = attn + (size_t)row * L_;

  __shared__ float wsum[8];
  float t[8];
  float s = 0.f;
#pragma unroll
  for (int kk = 0; kk < 8; ++kk) {
    int j = threadIdx.x + kk * 256;
    float val = mrow[j] ? 0.f : eb[j];
    t[kk] = val;
    s += val;
  }
#pragma unroll
  for (int off = 16; off > 0; off >>= 1)
    s += __shfl_xor(s, off, 32);
  if ((threadIdx.x & 31) == 0) wsum[threadIdx.x >> 5] = s;
  __syncthreads();
  float Z = 0.f;
#pragma unroll
  for (int w = 0; w < 8; ++w) Z += wsum[w];
  const float invZ = 1.0f / Z;
  if (abf) {
    __bf16* ab = abf + (size_t)row * L_;
#pragma unroll
    for (int kk = 0; kk < 8; ++kk) {
      int j = threadIdx.x + kk * 256;
      float val = t[kk] * invZ;
      arow[j] = val;
      ab[j] = (__bf16)val;
    }
  } else {
#pragma unroll
    for (int kk = 0; kk < 8; ++kk) {
      int j = threadIdx.x + kk * 256;
      arow[j] = t[kk] * invZ;
    }
  }
}

// ---------------------------------------------------------------------------
// Kernel 2b: vt[b,d,j] = bf16(v[b,j,d])   (one-shot transpose + convert so the
// GEMM's B tiles are contiguous 64B rows ready for async LDS copies)
// Tile 64(j) x 64(d) via LDS.
// ---------------------------------------------------------------------------
__global__ __launch_bounds__(256)
void v_to_bf16_t_kernel(const float* __restrict__ v, __bf16* __restrict__ vt) {
  __shared__ __attribute__((aligned(16))) __bf16 t[64][72];
  const int b  = blockIdx.z;
  const int j0 = blockIdx.y * 64;
  const int d0 = blockIdx.x * 64;
  const float* vb = v + (size_t)b * L_ * D_;
  __bf16* vtb = vt + (size_t)b * D_ * L_;

  const int dd = (threadIdx.x & 15) * 4;           // 0..60
#pragma unroll
  for (int it = 0; it < 4; ++it) {
    int jj = (threadIdx.x >> 4) + it * 16;         // 0..63
    float4 f = *(const float4*)(vb + (size_t)(j0 + jj) * D_ + d0 + dd);
    t[dd + 0][jj] = (__bf16)f.x;
    t[dd + 1][jj] = (__bf16)f.y;
    t[dd + 2][jj] = (__bf16)f.z;
    t[dd + 3][jj] = (__bf16)f.w;
  }
  __syncthreads();
  const int c = threadIdx.x & 7;                   // 8-elem chunk within row
#pragma unroll
  for (int it = 0; it < 2; ++it) {
    int d = (threadIdx.x >> 3) + it * 32;          // 0..63
    v8bf val = *(const v8bf*)&t[d][c * 8];
    *(v8bf*)(vtb + (size_t)(d0 + d) * L_ + j0 + c * 8) = val;
  }
}

// ---------------------------------------------------------------------------
// Kernel 3 (fast path): out[b] = attn_bf16[b] @ v_bf16[b], f32 accumulate.
// 256 threads = 8 waves; block tile 128x128; wave tile 32x64 = 2x4 WMMA frags.
// Double-buffered LDS fed by global_load_async_to_lds_b128 (ASYNCcnt pipelined).
// Fully uniform K-loop: every iteration prefetches tile min(k0+32, L-32) into
// the spare buffer (last iteration redundantly re-fetches itself), so there is
// a single compute instantiation -> accumulators stay in place, no epilogue.
// ---------------------------------------------------------------------------
__global__ __launch_bounds__(256)
void attn_v_gemm_bf16_kernel(const __bf16* __restrict__ abf,
                             const __bf16* __restrict__ vt,
                             float* __restrict__ out) {
  __shared__ __attribute__((aligned(16))) __bf16 As[2][128][40];
  __shared__ __attribute__((aligned(16))) __bf16 Bs[2][128][40];

  const int tid  = threadIdx.x;
  const int lane = tid & 31;
  const int wid  = tid >> 5;
  const int wr   = wid >> 1;                 // 0..3 -> rows wr*32
  const int wc   = wid & 1;                  // 0..1 -> cols wc*64

  const int bz      = blockIdx.z;            // batch
  const int rowBase = blockIdx.y * 128;      // within L
  const int colBase = blockIdx.x * 128;      // within D

  const __bf16* Ab  = abf + (size_t)bz * L_ * L_ + (size_t)rowBase * L_;
  const __bf16* Vtb = vt  + (size_t)bz * D_ * L_ + (size_t)colBase * L_;

  const v8f zacc = {0.f, 0.f, 0.f, 0.f, 0.f, 0.f, 0.f, 0.f};
  v8f acc[2][4];
#pragma unroll
  for (int i = 0; i < 2; ++i)
#pragma unroll
    for (int j = 0; j < 4; ++j) acc[i][j] = zacc;

  const int m  = lane & 15;
  const int kh = lane >> 4;

  // per-thread copy slots: 2 x (row, 16B-chunk) for each of A and B
  const int r0 = tid >> 2;                   // 0..63
  const int c0 = tid & 3;
  const int r1 = r0 + 64;                    // 64..127
  const int c1 = c0;

  // issue one 128x32 bf16 tile pair (A,B) into buffer `buf`: 4 async b128/thread
  auto issue_tiles = [&](int k0, int buf) {
    async_copy_b128((unsigned int)(uintptr_t)&As[buf][r0][c0 * 8],
                    Ab + (size_t)r0 * L_ + k0 + c0 * 8);
    async_copy_b128((unsigned int)(uintptr_t)&Bs[buf][r0][c0 * 8],
                    Vtb + (size_t)r0 * L_ + k0 + c0 * 8);
    async_copy_b128((unsigned int)(uintptr_t)&As[buf][r1][c1 * 8],
                    Ab + (size_t)r1 * L_ + k0 + c1 * 8);
    async_copy_b128((unsigned int)(uintptr_t)&Bs[buf][r1][c1 * 8],
                    Vtb + (size_t)r1 * L_ + k0 + c1 * 8);
  };

  issue_tiles(0, 0);
  int cur = 0;
  for (int k0 = 0; k0 < L_; k0 += 32) {
    const int knext = (k0 + 32 < L_) ? (k0 + 32) : k0;  // s_cselect, no branch
    issue_tiles(knext, cur ^ 1);
    asm volatile("s_wait_asynccnt 0x4" ::: "memory");   // current tiles resident
    __syncthreads();

    v16bf afrag[2], bfrag[4];
#pragma unroll
    for (int i = 0; i < 2; ++i) {
      const int ar = wr * 32 + i * 16 + m;
      v8bf lo = *(const v8bf*)&As[cur][ar][kh * 8];        // K = 8*kh..+7
      v8bf hi = *(const v8bf*)&As[cur][ar][16 + kh * 8];   // K = 16+8*kh..
      afrag[i] = __builtin_shufflevector(lo, hi, 0, 1, 2, 3, 4, 5, 6, 7,
                                         8, 9, 10, 11, 12, 13, 14, 15);
    }
#pragma unroll
    for (int j = 0; j < 4; ++j) {
      const int bc = wc * 64 + j * 16 + m;
      v8bf lo = *(const v8bf*)&Bs[cur][bc][kh * 8];        // K = 8*kh..+7
      v8bf hi = *(const v8bf*)&Bs[cur][bc][16 + kh * 8];   // K = 16+8*kh..
      bfrag[j] = __builtin_shufflevector(lo, hi, 0, 1, 2, 3, 4, 5, 6, 7,
                                         8, 9, 10, 11, 12, 13, 14, 15);
    }
#pragma unroll
    for (int i = 0; i < 2; ++i)
#pragma unroll
      for (int j = 0; j < 4; ++j)
        acc[i][j] = __builtin_amdgcn_wmma_f32_16x16x32_bf16(
            false, afrag[i], false, bfrag[j], (short)0, acc[i][j], false, false);

    __syncthreads();                                     // reads done before reuse
    cur ^= 1;
  }
  // (up to 4 async loads still outstanding; drained by S_ENDPGM's wait-idle)

  const int n  = lane & 15;
  const int h8 = (lane >> 4) * 8;
  float* outB = out + (size_t)bz * L_ * D_;
#pragma unroll
  for (int i = 0; i < 2; ++i) {
    const int rb = rowBase + wr * 32 + i * 16 + h8;
#pragma unroll
    for (int j = 0; j < 4; ++j) {
      const int cb = colBase + wc * 64 + j * 16 + n;
      float* op = outB + (size_t)rb * D_ + cb;
#pragma unroll
      for (int r = 0; r < 8; ++r)
        op[(size_t)r * D_] = acc[i][j][r];
    }
  }
}

// ---------------------------------------------------------------------------
// Kernel 3 (fallback, small-ws): same GEMM but reading f32 attn/v and
// converting to bf16 on the way into LDS.
// ---------------------------------------------------------------------------
__global__ __launch_bounds__(256)
void attn_v_gemm_f32_kernel(const float* __restrict__ attn,
                            const float* __restrict__ v,
                            float* __restrict__ out) {
  __shared__ __attribute__((aligned(16))) __bf16 As[128][40];
  __shared__ __attribute__((aligned(16))) __bf16 Bs[128][40];

  const int tid  = threadIdx.x;
  const int lane = tid & 31;
  const int wid  = tid >> 5;
  const int wr   = wid >> 1;
  const int wc   = wid & 1;

  const int bz      = blockIdx.z;
  const int rowBase = blockIdx.y * 128;
  const int colBase = blockIdx.x * 128;

  const float* Abase = attn + (size_t)bz * L_ * L_;
  const float* Vbase = v    + (size_t)bz * L_ * D_;

  const v8f zacc = {0.f, 0.f, 0.f, 0.f, 0.f, 0.f, 0.f, 0.f};
  v8f acc[2][4];
#pragma unroll
  for (int i = 0; i < 2; ++i)
#pragma unroll
    for (int j = 0; j < 4; ++j) acc[i][j] = zacc;

  const int m  = lane & 15;
  const int kh = lane >> 4;

  for (int k0 = 0; k0 < L_; k0 += 32) {
    {
      const int c4 = (tid & 7) * 4;
#pragma unroll
      for (int it = 0; it < 4; ++it) {
        const int r = (tid >> 3) + it * 32;
        float4 f = *(const float4*)(Abase + (size_t)(rowBase + r) * L_ + k0 + c4);
        As[r][c4 + 0] = (__bf16)f.x;
        As[r][c4 + 1] = (__bf16)f.y;
        As[r][c4 + 2] = (__bf16)f.z;
        As[r][c4 + 3] = (__bf16)f.w;
      }
    }
    {
      const int d4 = (tid & 31) * 4;
#pragma unroll
      for (int it = 0; it < 4; ++it) {
        const int j = (tid >> 5) + it * 8;
        float4 f = *(const float4*)(Vbase + (size_t)(k0 + j) * D_ + colBase + d4);
        Bs[d4 + 0][j] = (__bf16)f.x;
        Bs[d4 + 1][j] = (__bf16)f.y;
        Bs[d4 + 2][j] = (__bf16)f.z;
        Bs[d4 + 3][j] = (__bf16)f.w;
      }
    }
    __syncthreads();

    v16bf afrag[2], bfrag[4];
#pragma unroll
    for (int i = 0; i < 2; ++i) {
      const int ar = wr * 32 + i * 16 + m;
      v8bf lo = *(const v8bf*)&As[ar][kh * 8];
      v8bf hi = *(const v8bf*)&As[ar][16 + kh * 8];
      afrag[i] = __builtin_shufflevector(lo, hi, 0, 1, 2, 3, 4, 5, 6, 7,
                                         8, 9, 10, 11, 12, 13, 14, 15);
    }
#pragma unroll
    for (int j = 0; j < 4; ++j) {
      const int bc = wc * 64 + j * 16 + m;
      v8bf lo = *(const v8bf*)&Bs[bc][kh * 8];
      v8bf hi = *(const v8bf*)&Bs[bc][16 + kh * 8];
      bfrag[j] = __builtin_shufflevector(lo, hi, 0, 1, 2, 3, 4, 5, 6, 7,
                                         8, 9, 10, 11, 12, 13, 14, 15);
    }
#pragma unroll
    for (int i = 0; i < 2; ++i)
#pragma unroll
      for (int j = 0; j < 4; ++j)
        acc[i][j] = __builtin_amdgcn_wmma_f32_16x16x32_bf16(
            false, afrag[i], false, bfrag[j], (short)0, acc[i][j], false, false);
    __syncthreads();
  }

  const int n  = lane & 15;
  const int h8 = (lane >> 4) * 8;
  float* outB = out + (size_t)bz * L_ * D_;
#pragma unroll
  for (int i = 0; i < 2; ++i) {
    const int rb = rowBase + wr * 32 + i * 16 + h8;
#pragma unroll
    for (int j = 0; j < 4; ++j) {
      const int cb = colBase + wc * 64 + j * 16 + n;
      float* op = outB + (size_t)rb * D_ + cb;
#pragma unroll
      for (int r = 0; r < 8; ++r)
        op[(size_t)r * D_] = acc[i][j][r];
    }
  }
}

// ---------------------------------------------------------------------------
// Launch
// ---------------------------------------------------------------------------
extern "C" void kernel_launch(void* const* d_in, const int* in_sizes, int n_in,
                              void* d_out, int out_size, void* d_ws, size_t ws_size,
                              hipStream_t stream) {
  // setup_inputs order: q, k, v, attn_mask, W1, b1, W2, b2
  const float*         k    = (const float*)d_in[1];
  const float*         v    = (const float*)d_in[2];
  const unsigned char* mask = (const unsigned char*)d_in[3];  // bool, 1 byte
  const float*         W2   = (const float*)d_in[6];
  const float*         b2   = (const float*)d_in[7];

  float* out  = (float*)d_out;                       // [B, L, D]
  float* attn = out + (size_t)B_ * L_ * D_;          // [B, L, L]

  const size_t eBytes   = (size_t)B_ * L_ * sizeof(float);           // 64 KB
  const size_t abfBytes = (size_t)B_ * L_ * L_ * sizeof(__bf16);     // 64 MB
  const size_t vtBytes  = (size_t)B_ * L_ * D_ * sizeof(__bf16);     // 32 MB

  float* e = (float*)d_ws;
  proj_exp_kernel<<<B_ * L_ / 8, 256, 0, stream>>>(k, W2, b2, e);

  if (ws_size >= eBytes + abfBytes + vtBytes) {
    // Fast path: bf16 operand copies + async-LDS double-buffered WMMA GEMM.
    __bf16* abf = (__bf16*)((char*)d_ws + eBytes);
    __bf16* vt  = (__bf16*)((char*)d_ws + eBytes + abfBytes);
    softmax_rows_kernel<<<B_ * L_, 256, 0, stream>>>(mask, e, attn, abf);
    v_to_bf16_t_kernel<<<dim3(D_ / 64, L_ / 64, B_), 256, 0, stream>>>(v, vt);
    attn_v_gemm_bf16_kernel<<<dim3(D_ / 128, L_ / 128, B_), 256, 0, stream>>>(abf, vt, out);
  } else {
    // Fallback: read f32 attn (from d_out) and f32 v, convert in-kernel.
    softmax_rows_kernel<<<B_ * L_, 256, 0, stream>>>(mask, e, attn, nullptr);
    attn_v_gemm_f32_kernel<<<dim3(D_ / 128, L_ / 128, B_), 256, 0, stream>>>(attn, v, out);
  }
}
